// IoTAnomalyDetectorLSTMEncDecNetImp_37014028157709
// MI455X (gfx1250) — matrix operand order repaired and checked by reading, not compiled, not run on hardware
//
#include <hip/hip_runtime.h>

// ---------------------------------------------------------------------------
// LSTM encoder-decoder (B=256, T=512, I=128, H=256) on gfx1250 (MI455X).
// bf16-staged activations/weights; per-timestep kernels are software-pipelined
// global_load_b128 -> v_wmma_f32_16x16x32_bf16 chains fused with the LSTM cell.
// Pipeline shape enforced with __builtin_amdgcn_sched_group_barrier.
// ---------------------------------------------------------------------------

typedef __attribute__((ext_vector_type(16))) __bf16    v16bf;
typedef __attribute__((ext_vector_type(8)))  float     v8f;
typedef __attribute__((ext_vector_type(4)))  unsigned  u4;

static constexpr int Bb = 256;   // batch
static constexpr int Tt = 512;   // sequence length
static constexpr int Ii = 128;   // input dim
static constexpr int Hh = 256;   // hidden dim

#define DEVFN static __device__ __forceinline__

#if __has_builtin(__builtin_amdgcn_sched_group_barrier)
#define SGB(mask, size, id) __builtin_amdgcn_sched_group_barrier((mask), (size), (id))
#else
#define SGB(mask, size, id) ((void)0)
#endif
// sched_group_barrier masks
#define SG_MFMA  0x008   // MFMA/WMMA
#define SG_VMRD  0x020   // VMEM read

// fp32 -> bf16, round-to-nearest-even (outside the hot chain only).
DEVFN __bf16 f2bf(float f) {
    unsigned u = __builtin_bit_cast(unsigned, f);
    u += 0x7FFFu + ((u >> 16) & 1u);
    unsigned short s = (unsigned short)(u >> 16);
    return __builtin_bit_cast(__bf16, s);
}

DEVFN float sigf(float x) { return 1.0f / (1.0f + __expf(-x)); }

DEVFN float tanh_fast(float x) {
    x = fminf(fmaxf(x, -15.0f), 15.0f);
    float e = __expf(2.0f * x);
    return (e - 1.0f) / (e + 1.0f);
}

// A fragment: 16x32 (MxK) bf16 row-major, leading dim `ld` (elements).
// Lane l holds row (l&15); lanes 0-15 carry K={k0..k0+7, k0+16..k0+23},
// lanes 16-31 the +8 chunks. Two b128 loads.
DEVFN v16bf load_a_bf(const __bf16* __restrict__ base, int ld, int lane, int k0) {
    const int row = lane & 15;
    const int kb  = k0 + ((lane >> 4) << 3);
    const __bf16* p = base + (size_t)row * (size_t)ld + kb;
    union { u4 q[2]; v16bf v; } u;
    u.q[0] = *(const u4*)(p);        // K kb .. kb+7
    u.q[1] = *(const u4*)(p + 16);   // K kb+16 .. kb+23
    return u.v;
}

// B fragment: 32x16 (KxN) bf16. Lane l holds column (l&15); lanes 0-15 carry
// K={k0..k0+7, k0+16..k0+23}, lanes 16-31 the +8 chunks. W stored [N,K] row-major so one
// result column is a contiguous 32-byte run. Two b128 loads.
DEVFN v16bf load_b_bf(const __bf16* __restrict__ wbase, int ld, int lane, int k0) {
    const int col = lane & 15;
    const int kb  = k0 + ((lane >> 4) << 3);
    const __bf16* p = wbase + (size_t)col * (size_t)ld + kb;
    union { u4 q[2]; v16bf v; } u;
    u.q[0] = *(const u4*)(p);        // K kb .. kb+7
    u.q[1] = *(const u4*)(p + 16);   // K kb+16 .. kb+23
    return u.v;
}

DEVFN v8f wmma_bf16(v16bf a, v16bf b, v8f c) {
    return __builtin_amdgcn_wmma_f32_16x16x32_bf16(
        /*neg_a=*/false, a, /*neg_b=*/false, b,
        /*c_mod=*/(short)0, c, /*reuse_a=*/false, /*reuse_b=*/false);
}

// ---------------------------------------------------------------------------
// One LSTM timestep, fully fused, software-pipelined (depth 2):
//   gates = X @ Wih^T + h_in @ Whh^T + (bih + bhh)
//   c     = sig(f)*c + sig(i)*tanh(g);  h_out = sig(o)*tanh(c)
// 256 waves: wave (m,n) computes the i/f/g/o 16x16 tiles at its position.
// ---------------------------------------------------------------------------
__global__ __launch_bounds__(256) void lstm_step_kernel(
    const __bf16* __restrict__ X, int ldx,
    const __bf16* __restrict__ Wih, const __bf16* __restrict__ Whh,
    const float*  __restrict__ bih, const float*  __restrict__ bhh,
    const __bf16* __restrict__ h_in, __bf16* __restrict__ h_out,
    float* __restrict__ c)
{
    const int tid  = blockIdx.x * blockDim.x + threadIdx.x;
    const int wid  = tid >> 5;          // 0..255
    const int lane = tid & 31;
    const int m0   = (wid >> 4) << 4;   // batch-row tile base
    const int n0   = (wid & 15) << 4;   // hidden-col tile base (within a gate)

    constexpr int KI = Ii / 32;         // 4  input K-steps
    constexpr int KT = KI + Hh / 32;    // 12 total K-steps

    const __bf16* Xb = X    + (size_t)m0 * (size_t)ldx;
    const __bf16* Hb = h_in + (size_t)m0 * Hh;

    v8f acc[4];
#pragma unroll
    for (int g = 0; g < 4; ++g)
#pragma unroll
        for (int j = 0; j < 8; ++j) acc[g][j] = 0.0f;

    // Stage s (compile-time under full unroll): s<KI -> input part, else
    // recurrent part. All selects fold to constants.
    auto loadA = [&](int s) -> v16bf {
        return (s < KI) ? load_a_bf(Xb, ldx, lane, s * 32)
                        : load_a_bf(Hb, Hh,  lane, (s - KI) * 32);
    };
    auto loadB = [&](int s, int g) -> v16bf {
        return (s < KI)
            ? load_b_bf(Wih + (size_t)(g * Hh + n0) * Ii, Ii, lane, s * 32)
            : load_b_bf(Whh + (size_t)(g * Hh + n0) * Hh, Hh, lane, (s - KI) * 32);
    };

    // Prologue: fragments for stage 0.
    v16bf aC  = loadA(0);
    v16bf bC0 = loadB(0, 0), bC1 = loadB(0, 1), bC2 = loadB(0, 2), bC3 = loadB(0, 3);

#pragma unroll
    for (int s = 0; s < KT; ++s) {
        v16bf aN, bN0, bN1, bN2, bN3;
        if (s + 1 < KT) {               // issue next stage's 10 loads first
            aN  = loadA(s + 1);
            bN0 = loadB(s + 1, 0);
            bN1 = loadB(s + 1, 1);
            bN2 = loadB(s + 1, 2);
            bN3 = loadB(s + 1, 3);
        }
        acc[0] = wmma_bf16(aC, bC0, acc[0]);
        acc[1] = wmma_bf16(aC, bC1, acc[1]);
        acc[2] = wmma_bf16(aC, bC2, acc[2]);
        acc[3] = wmma_bf16(aC, bC3, acc[3]);
        if (s + 1 < KT) {
            aC = aN; bC0 = bN0; bC1 = bN1; bC2 = bN2; bC3 = bN3;
        }
    }

    // Per-lane column index (all 8 accumulator rows share one N).
    const int nc = n0 + (lane & 15);
    float bg[4];
#pragma unroll
    for (int g = 0; g < 4; ++g) bg[g] = bih[g * Hh + nc] + bhh[g * Hh + nc];

    const int mbase = m0 + ((lane >> 4) << 3);
    float cv[8];
#pragma unroll
    for (int j = 0; j < 8; ++j)
        cv[j] = c[(size_t)(mbase + j) * Hh + nc];

#pragma unroll
    for (int j = 0; j < 8; ++j) {
        const size_t idx = (size_t)(mbase + j) * Hh + nc;
        float iv = sigf(acc[0][j] + bg[0]);
        float fv = sigf(acc[1][j] + bg[1]);
        float gv = tanh_fast(acc[2][j] + bg[2]);
        float ov = sigf(acc[3][j] + bg[3]);
        float cn = fv * cv[j] + iv * gv;
        c[idx]     = cn;
        h_out[idx] = f2bf(ov * tanh_fast(cn));
    }

    // -------- pipeline shape (one scheduling region; loop fully unrolled) ----
    // [VMEM x20: stages 0+1] then 10x [WMMA x4 | VMEM x10: stage s+2],
    // then the final two stages' WMMAs, then the 16 epilogue reads (bias + c).
    SGB(SG_VMRD, 20, 0);
#pragma unroll
    for (int s = 0; s < KT - 2; ++s) {
        SGB(SG_MFMA, 4, 0);
        SGB(SG_VMRD, 10, 0);
    }
    SGB(SG_MFMA, 4, 0);
    SGB(SG_MFMA, 4, 0);
    SGB(SG_VMRD, 16, 0);
}

// ---------------------------------------------------------------------------
// pred = sigmoid(h @ linW^T + linb); writes bf16 feedback + fp32 out[:, t, :].
// 128 waves: 16 row tiles x 8 col tiles (I = 128). Depth-2 pipelined.
// ---------------------------------------------------------------------------
__global__ __launch_bounds__(256) void linear_sigmoid_kernel(
    const __bf16* __restrict__ Hc, const __bf16* __restrict__ linW,
    const float*  __restrict__ linb, __bf16* __restrict__ pred,
    float* __restrict__ out, int t)
{
    const int tid  = blockIdx.x * blockDim.x + threadIdx.x;
    const int wid  = tid >> 5;          // 0..127
    const int lane = tid & 31;
    const int m0   = (wid >> 3) << 4;
    const int n0   = (wid & 7) << 4;

    constexpr int KT = Hh / 32;         // 8 K-steps

    v8f acc;
#pragma unroll
    for (int j = 0; j < 8; ++j) acc[j] = 0.0f;

    const __bf16* Hb = Hc   + (size_t)m0 * Hh;
    const __bf16* Wb = linW + (size_t)n0 * Hh;

    v16bf aC = load_a_bf(Hb, Hh, lane, 0);
    v16bf bC = load_b_bf(Wb, Hh, lane, 0);
#pragma unroll
    for (int s = 0; s < KT; ++s) {
        v16bf aN, bN;
        if (s + 1 < KT) {
            aN = load_a_bf(Hb, Hh, lane, (s + 1) * 32);
            bN = load_b_bf(Wb, Hh, lane, (s + 1) * 32);
        }
        acc = wmma_bf16(aC, bC, acc);
        if (s + 1 < KT) { aC = aN; bC = bN; }
    }

    const int nc = n0 + (lane & 15);
    const float bb = linb[nc];
    const int mbase = m0 + ((lane >> 4) << 3);
#pragma unroll
    for (int j = 0; j < 8; ++j) {
        const int m = mbase + j;
        float v = sigf(acc[j] + bb);
        pred[(size_t)m * Ii + nc] = f2bf(v);
        out[((size_t)m * Tt + t) * Ii + nc] = v;
    }

    // [VMEM x8: stages 0+1] then 6x [WMMA | VMEM x4], final 2 WMMAs, bias read.
    SGB(SG_VMRD, 8, 0);
#pragma unroll
    for (int s = 0; s < KT - 2; ++s) {
        SGB(SG_MFMA, 1, 0);
        SGB(SG_VMRD, 4, 0);
    }
    SGB(SG_MFMA, 1, 0);
    SGB(SG_MFMA, 1, 0);
    SGB(SG_VMRD, 1, 0);
}

// fp32 -> bf16 bulk conversion (weights + whole input sequence), once/launch.
__global__ void cvt_bf16_kernel(const float* __restrict__ src,
                                __bf16* __restrict__ dst, int n) {
    int i = blockIdx.x * blockDim.x + threadIdx.x;
    if (i < n) dst[i] = f2bf(src[i]);
}

__global__ void init_state_kernel(__bf16* __restrict__ h, float* __restrict__ c) {
    int i = blockIdx.x * blockDim.x + threadIdx.x;
    if (i < Bb * Hh) {
        h[i] = __builtin_bit_cast(__bf16, (unsigned short)0);
        c[i] = 0.0f;
    }
}

extern "C" void kernel_launch(void* const* d_in, const int* in_sizes, int n_in,
                              void* d_out, int out_size, void* d_ws, size_t ws_size,
                              hipStream_t stream) {
    (void)in_sizes; (void)n_in; (void)out_size; (void)ws_size;

    const float* x     = (const float*)d_in[0];   // [B, T, I]
    const float* eWih  = (const float*)d_in[1];   // [4H, I]
    const float* eWhh  = (const float*)d_in[2];   // [4H, H]
    const float* ebih  = (const float*)d_in[3];   // [4H]
    const float* ebhh  = (const float*)d_in[4];   // [4H]
    const float* dWih  = (const float*)d_in[5];   // [4H, I]
    const float* dWhh  = (const float*)d_in[6];   // [4H, H]
    const float* dbih  = (const float*)d_in[7];   // [4H]
    const float* dbhh  = (const float*)d_in[8];   // [4H]
    const float* linW  = (const float*)d_in[9];   // [I, H]
    const float* linb  = (const float*)d_in[10];  // [I]
    float* out = (float*)d_out;                   // [B, T, I]

    // ---- workspace layout ----
    __bf16* p = (__bf16*)d_ws;
    __bf16* xbf    = p; p += (size_t)Bb * Tt * Ii;   // 16.7M elems (32 MB)
    __bf16* eWihB  = p; p += 4 * Hh * Ii;
    __bf16* eWhhB  = p; p += 4 * Hh * Hh;
    __bf16* dWihB  = p; p += 4 * Hh * Ii;
    __bf16* dWhhB  = p; p += 4 * Hh * Hh;
    __bf16* linWB  = p; p += Ii * Hh;
    __bf16* hA     = p; p += Bb * Hh;
    __bf16* hB     = p; p += Bb * Hh;
    __bf16* pred   = p; p += Bb * Ii;
    float*  cS     = (float*)p;                      // [B, H] fp32

    // ---- one-time bf16 conversions ----
    {
        const int nx = Bb * Tt * Ii;
        cvt_bf16_kernel<<<(nx + 255) / 256, 256, 0, stream>>>(x, xbf, nx);
        cvt_bf16_kernel<<<(4 * Hh * Ii + 255) / 256, 256, 0, stream>>>(eWih, eWihB, 4 * Hh * Ii);
        cvt_bf16_kernel<<<(4 * Hh * Hh + 255) / 256, 256, 0, stream>>>(eWhh, eWhhB, 4 * Hh * Hh);
        cvt_bf16_kernel<<<(4 * Hh * Ii + 255) / 256, 256, 0, stream>>>(dWih, dWihB, 4 * Hh * Ii);
        cvt_bf16_kernel<<<(4 * Hh * Hh + 255) / 256, 256, 0, stream>>>(dWhh, dWhhB, 4 * Hh * Hh);
        cvt_bf16_kernel<<<(Ii * Hh + 255) / 256, 256, 0, stream>>>(linW, linWB, Ii * Hh);
    }

    __bf16* hbuf[2] = {hA, hB};
    init_state_kernel<<<(Bb * Hh + 255) / 256, 256, 0, stream>>>(hA, cS);

    // ---- encoder: T steps; input row stride is T*I within xbf ----
    int par = 0;
    for (int t = 0; t < Tt; ++t) {
        lstm_step_kernel<<<32, 256, 0, stream>>>(
            xbf + (size_t)t * Ii, Tt * Ii,
            eWihB, eWhhB, ebih, ebhh,
            hbuf[par], hbuf[par ^ 1], cS);
        par ^= 1;
    }

    // prediction at last timestep from encoder final hidden state
    linear_sigmoid_kernel<<<16, 256, 0, stream>>>(
        hbuf[par], linWB, linb, pred, out, Tt - 1);

    // ---- decoder: T-1 steps backwards in time, feeding back pred ----
    for (int k = 0; k < Tt - 1; ++k) {
        lstm_step_kernel<<<32, 256, 0, stream>>>(
            pred, Ii,
            dWihB, dWhhB, dbih, dbhh,
            hbuf[par], hbuf[par ^ 1], cS);
        par ^= 1;
        linear_sigmoid_kernel<<<16, 256, 0, stream>>>(
            hbuf[par], linWB, linb, pred, out, Tt - 2 - k);
    }
}
